// DUF_Wguide_resolutionRGB_33354716021613
// MI455X (gfx1250) — matrix operand order repaired and checked by reading, not compiled, not run on hardware
//
#include <hip/hip_runtime.h>
#include <hip/hip_bf16.h>
#include <cstdint>
#include <cstddef>

// ---------------------------------------------------------------------------
// CDNA5 (gfx1250) implementation of the DUF block stack.
// GEMM-shaped convs run on v_wmma_f32_16x16x32_f16, 8 waves per block.
// Weight (B) fragments are staged block-cooperatively into LDS (ds_load_b128),
// activations stream from global as 2x b128 per lane per k-step.
// 3x3 convs = 9 shifted K-deep GEMMs accumulated in the f32 WMMA accumulator.
// ---------------------------------------------------------------------------

typedef __attribute__((ext_vector_type(16))) _Float16 v16h;
typedef __attribute__((ext_vector_type(8)))  _Float16 v8h;
typedef __attribute__((ext_vector_type(8)))  float    v8f;

union HFrag { v16h v; v8h h[2]; };

static __device__ __forceinline__ v16h frag_zero() { v16h z = {}; return z; }

// A fragment: 16-bit A 16x32 layout (ISA 7.12.2).
// lane<16 : row M=lane,    K = {k0..k0+7} in v0..3, {k0+16..k0+23} in v4..7
// lane>=16: row M=lane-16, K = {k0+8..k0+15},        {k0+24..k0+31}
static __device__ __forceinline__ v16h load_frag_a(const _Float16* __restrict__ row,
                                                   int k0, int hi) {
  HFrag a;
  a.h[0] = *(const v8h*)(row + k0 + hi * 8);
  a.h[1] = *(const v8h*)(row + k0 + 16 + hi * 8);
  return a.v;
}

// B fragment: 16-bit B 32x16. lane<16: col N=lane, K={0..7,16..23};
// lane>=16: col N=lane-16, K={8..15,24..31}. Row (= output channel) has K contiguous.
static __device__ __forceinline__ v16h load_frag_b(const _Float16* __restrict__ wrow,
                                                   int k0, int hi) {
  HFrag b;
  b.h[0] = *(const v8h*)(wrow + k0 + hi * 8);
  b.h[1] = *(const v8h*)(wrow + k0 + 16 + hi * 8);
  return b.v;
}

static __device__ __forceinline__ v8f wmma_f16(v16h a, v16h b, v8f c) {
  // D = A(16x32 f16) * B(32x16 f16) + C(16x16 f32)
  return __builtin_amdgcn_wmma_f32_16x16x32_f16(
      /*neg_a=*/false, a, /*neg_b=*/false, b,
      /*c_mod=*/(short)0, c, /*reuse_a=*/false, /*reuse_b=*/false);
}

// C/D 16x16 f32 layout: lane<16 -> N=lane, M=0..7 in v0..7; lane>=16 -> N=lane-16, M=8..15.
template <int NB, bool RELU, bool OUT16>
static __device__ __forceinline__ void store_tile(const v8f* acc, long pix0, int n0,
                                                  int m, int hi,
                                                  const float* __restrict__ bias,
                                                  int Cout, int ldOut,
                                                  void* __restrict__ outp) {
#pragma unroll
  for (int j = 0; j < NB; ++j) {
    int n = n0 + j * 16 + m;
    if (n < Cout) {
      float bi = bias ? bias[n] : 0.f;
#pragma unroll
      for (int r = 0; r < 8; ++r) {
        float v = acc[j][r] + bi;
        if (RELU) v = fmaxf(v, 0.f);
        long pix = pix0 + hi * 8 + r;
        if (OUT16) ((_Float16*)outp)[pix * (long)ldOut + n] = (_Float16)v;
        else       ((float*)outp)[pix * (long)ldOut + n] = v;
      }
    }
  }
}

// ------------------------- 1x1 conv as WMMA GEMM ---------------------------
// Block = 256 threads = 8 waves; each wave owns one 16-pixel tile.
// The [NB*16][K] weight slice for this blockIdx.y is staged into LDS once,
// then all 8 waves feed their B fragments from LDS (rows padded +8 halves).
template <int NB, bool RELU, bool OUT16>
__global__ void __launch_bounds__(256)
conv1x1_wmma(const _Float16* __restrict__ act, int ldA, int K,
             const _Float16* __restrict__ wT, const float* __restrict__ bias,
             int Cout, int ldOut, void* __restrict__ outp) {
  extern __shared__ _Float16 smem[];
  const int tid = threadIdx.x;
  const int n0 = blockIdx.y * (NB * 16);
  const int ldS = K + 8;            // pad rows to de-conflict LDS banks
  const int KV = K >> 3;            // v8h chunks per row

  for (int idx = tid; idx < NB * 16 * KV; idx += 256) {
    const int row = idx / KV, ch = (idx % KV) * 8;
    *(v8h*)(smem + (long)row * ldS + ch) = *(const v8h*)(wT + (long)(n0 + row) * K + ch);
  }
  __syncthreads();

  const int lane = tid & 31;
  const int wv = tid >> 5;
  const int m = lane & 15, hi = lane >> 4;
  const long pix0 = ((long)blockIdx.x * 8 + wv) * 16;
  const _Float16* arow = act + (pix0 + m) * (long)ldA;

  v8f zf = {};
  v8f acc[NB];
#pragma unroll
  for (int j = 0; j < NB; ++j) acc[j] = zf;

  for (int k0 = 0; k0 < K; k0 += 32) {
    v16h a = load_frag_a(arow, k0, hi);
#pragma unroll
    for (int j = 0; j < NB; ++j) {
      const _Float16* brow = smem + (long)(j * 16 + m) * ldS;
      v16h b = load_frag_b(brow, k0, hi);
      acc[j] = wmma_f16(a, b, acc[j]);
    }
  }
  store_tile<NB, RELU, OUT16>(acc, pix0, n0, m, hi, bias, Cout, ldOut, outp);
}

// ------------------ 3x3 conv as 9 shifted WMMA GEMMs -----------------------
// Per tap: stage that tap's [NB*16][K] weight slice to LDS, barrier, GEMM.
// Pixel tile = 16 consecutive x in one image row; 8 tiles (waves) per block.
template <int NB, bool RELU, bool OUT16>
__global__ void __launch_bounds__(256)
conv3x3_wmma(const _Float16* __restrict__ act, int ldA, int K,
             const _Float16* __restrict__ wT9, const float* __restrict__ bias,
             int Cout, int CoutPad, int ldOut, void* __restrict__ outp,
             int Himg, int Wimg) {
  extern __shared__ _Float16 smem[];
  const int tid = threadIdx.x;
  const int lane = tid & 31;
  const int wv = tid >> 5;
  const int m = lane & 15, hi = lane >> 4;
  const int tilesPerRow = Wimg / 16;
  const int n0 = blockIdx.y * (NB * 16);
  const int ldS = K + 8;
  const int KV = K >> 3;

  long t = (long)blockIdx.x * 8 + wv;
  const int wt = (int)(t % tilesPerRow); t /= tilesPerRow;
  const int h = (int)(t % Himg);
  const int nimg = (int)(t / Himg);
  const int x = wt * 16 + m;                 // this lane's pixel column

  v8f zf = {};
  v8f acc[NB];
#pragma unroll
  for (int j = 0; j < NB; ++j) acc[j] = zf;

  for (int tap = 0; tap < 9; ++tap) {
    // cooperative stage of this tap's weight slice
    for (int idx = tid; idx < NB * 16 * KV; idx += 256) {
      const int row = idx / KV, ch = (idx % KV) * 8;
      *(v8h*)(smem + (long)row * ldS + ch) =
          *(const v8h*)(wT9 + ((long)tap * CoutPad + n0 + row) * K + ch);
    }
    __syncthreads();

    const int dy = tap / 3 - 1, dx = tap % 3 - 1;
    const int hh = h + dy;
    const bool hok = (hh >= 0) & (hh < Himg);
    const int xx = x + dx;
    const bool ok = hok & (xx >= 0) & (xx < Wimg);
    const _Float16* arow =
        act + ((long)(nimg * Himg + (hok ? hh : 0)) * Wimg + (ok ? xx : 0)) * (long)ldA;

    for (int k0 = 0; k0 < K; k0 += 32) {
      v16h a = frag_zero();
      if (ok) a = load_frag_a(arow, k0, hi);   // zero-fill halo lanes; WMMA runs full-EXEC
#pragma unroll
      for (int j = 0; j < NB; ++j) {
        const _Float16* brow = smem + (long)(j * 16 + m) * ldS;
        v16h b = load_frag_b(brow, k0, hi);
        acc[j] = wmma_f16(a, b, acc[j]);
      }
    }
    __syncthreads();   // protect LDS before next tap's staging
  }
  const long pix0 = ((long)nimg * Himg + h) * Wimg + wt * 16;
  store_tile<NB, RELU, OUT16>(acc, pix0, n0, m, hi, bias, Cout, ldOut, outp);
}

// ------------------------------ BatchNorm ----------------------------------
__global__ void bn_stats(const _Float16* __restrict__ act, int ldA, int C, long P,
                         float* __restrict__ sum, float* __restrict__ sumsq) {
  const int c = threadIdx.x;
  if (c >= C) return;
  const long chunk = (P + gridDim.x - 1) / gridDim.x;
  long p0 = (long)blockIdx.x * chunk;
  long p1 = p0 + chunk; if (p1 > P) p1 = P;
  float s = 0.f, s2 = 0.f;
  for (long p = p0; p < p1; ++p) {
    float v = (float)act[p * (long)ldA + c];
    s += v; s2 += v * v;
  }
  atomicAdd(&sum[c], s);
  atomicAdd(&sumsq[c], s2);
}

__global__ void bn_apply_relu(const _Float16* __restrict__ act, int ldA, int C, long P,
                              const float* __restrict__ sum, const float* __restrict__ sumsq,
                              _Float16* __restrict__ out, int ldOut) {
  const long idx = (long)blockIdx.x * blockDim.x + threadIdx.x;
  const long total = P * (long)C;
  if (idx >= total) return;
  const int c = (int)(idx % C);
  const long p = idx / C;
  const float invP = 1.f / (float)P;
  const float mean = sum[c] * invP;
  const float var = sumsq[c] * invP - mean * mean;
  const float sc = rsqrtf(var + 1e-5f);
  const float v = ((float)act[p * (long)ldA + c] - mean) * sc;
  out[p * (long)ldOut + c] = (_Float16)fmaxf(v, 0.f);
}

// ------------------------- small direct kernels ----------------------------
// W (4,1,512,512) f32 -> wg [4][256][256][4] f16 (space-to-depth)
__global__ void s2d_kernel(const float* __restrict__ Wfull, _Float16* __restrict__ wg,
                           int Himg, int Wimg) {
  const long idx = (long)blockIdx.x * blockDim.x + threadIdx.x;
  const long total = 4L * Himg * Wimg * 4;
  if (idx >= total) return;
  const int c = (int)(idx & 3);
  long p = idx >> 2;
  const int x = (int)(p % Wimg); p /= Wimg;
  const int y = (int)(p % Himg);
  const int n = (int)(p / Himg);
  const long src = ((long)n * (2 * Himg) + (2 * y + (c >> 1))) * (2 * Wimg) + (2 * x + (c & 1));
  wg[idx] = (_Float16)Wfull[src];
}

// pre conv: 3x3, 4->64, direct VALU (tiny). Writes f16 into X[:, 0:64] (ld=128).
__global__ void conv_pre_kernel(const _Float16* __restrict__ wg,
                                const float* __restrict__ w,   // [64][4][3][3]
                                const float* __restrict__ b,
                                _Float16* __restrict__ X, int Himg, int Wimg) {
  const long gid = (long)blockIdx.x * blockDim.x + threadIdx.x;
  const int co = (int)(gid & 63);
  const long p = gid >> 6;
  const long P = 4L * Himg * Wimg;
  if (p >= P) return;
  const int x = (int)(p % Wimg);
  long t = p / Wimg;
  const int y = (int)(t % Himg);
  const int n = (int)(t / Himg);
  float acc = b[co];
  for (int dy = -1; dy <= 1; ++dy) {
    const int yy = y + dy;
    if (yy < 0 || yy >= Himg) continue;
    for (int dx = -1; dx <= 1; ++dx) {
      const int xx = x + dx;
      if (xx < 0 || xx >= Wimg) continue;
      const _Float16* ip = wg + ((long)(n * Himg + yy) * Wimg + xx) * 4;
#pragma unroll
      for (int i = 0; i < 4; ++i)
        acc += (float)ip[i] * w[((co * 4 + i) * 3 + (dy + 1)) * 3 + (dx + 1)];
    }
  }
  X[p * 128 + co] = (_Float16)acc;
}

// r2: 1x1, 128->4, direct (tiny)
__global__ void conv_r2_kernel(const _Float16* __restrict__ act,   // [P][128]
                               const float* __restrict__ w,        // [4][128]
                               const float* __restrict__ b,
                               float* __restrict__ out, long P) {  // [P][4]
  const long gid = (long)blockIdx.x * blockDim.x + threadIdx.x;
  const int co = (int)(gid & 3);
  const long p = gid >> 2;
  if (p >= P) return;
  float acc = b[co];
  const _Float16* a = act + p * 128;
  const float* wr = w + co * 128;
  for (int k = 0; k < 128; ++k) acc += (float)a[k] * wr[k];
  out[p * 4 + co] = acc;
}

// softmax over 25 taps for each (pixel, rr) — filt [P][100], channel = k*4 + r
__global__ void softmax25_kernel(_Float16* __restrict__ f, long P) {
  const long gid = (long)blockIdx.x * blockDim.x + threadIdx.x;
  const int r = (int)(gid & 3);
  const long p = gid >> 2;
  if (p >= P) return;
  _Float16* fp = f + p * 100;
  float v[25];
  float mx = -1e30f;
#pragma unroll
  for (int k = 0; k < 25; ++k) { v[k] = (float)fp[k * 4 + r]; mx = fmaxf(mx, v[k]); }
  float s = 0.f;
#pragma unroll
  for (int k = 0; k < 25; ++k) { v[k] = __expf(v[k] - mx); s += v[k]; }
  const float inv = 1.f / s;
#pragma unroll
  for (int k = 0; k < 25; ++k) fp[k * 4 + r] = (_Float16)(v[k] * inv);
}

// dyn_filter + pixel_shuffle(t) + pixel_shuffle(r) + add -> out[:, ci, 512, 512]
__global__ void duf_out_kernel(const float* __restrict__ bggr, int ci,
                               const _Float16* __restrict__ filt,  // [P][100]
                               const float* __restrict__ r2o,      // [P][4]
                               float* __restrict__ out, int Himg, int Wimg) {
  const long p = (long)blockIdx.x * blockDim.x + threadIdx.x;
  const long P = 4L * Himg * Wimg;
  if (p >= P) return;
  const int x = (int)(p % Wimg);
  long t = p / Wimg;
  const int y = (int)(t % Himg);
  const int n = (int)(t / Himg);
  const float* src = bggr + ((long)(n * 4 + ci) * Himg) * Wimg;
  float patch[25];
#pragma unroll
  for (int dy = -2; dy <= 2; ++dy) {
    const int yy = y + dy;
#pragma unroll
    for (int dx = -2; dx <= 2; ++dx) {
      const int xx = x + dx;
      patch[(dy + 2) * 5 + (dx + 2)] =
          (yy >= 0 && yy < Himg && xx >= 0 && xx < Wimg) ? src[(long)yy * Wimg + xx] : 0.f;
    }
  }
  const _Float16* fp = filt + p * 100;
  float res[4];
#pragma unroll
  for (int r = 0; r < 4; ++r) {
    float s = 0.f;
#pragma unroll
    for (int k = 0; k < 25; ++k) s += patch[k] * (float)fp[k * 4 + r];
    res[r] = s + r2o[p * 4 + r];
  }
  float* ob = out + ((long)(n * 4 + ci) * (2 * Himg)) * (2 * Wimg);
#pragma unroll
  for (int a = 0; a < 2; ++a)
#pragma unroll
    for (int b2 = 0; b2 < 2; ++b2)
      ob[(long)(2 * y + a) * (2 * Wimg) + (2 * x + b2)] = res[2 * a + b2];
}

// -------------------------- weight preprocessing ---------------------------
// 1x1 OIHW [O][I] f32 -> [Opad][I] f16 (zero-pad extra rows)
__global__ void prep_w1x1_kernel(const float* __restrict__ w, _Float16* __restrict__ o,
                                 int O, int Opad, int I) {
  const long idx = (long)blockIdx.x * blockDim.x + threadIdx.x;
  const long tot = (long)Opad * I;
  if (idx >= tot) return;
  const int k = (int)(idx % I);
  const int oo = (int)(idx / I);
  o[idx] = (oo < O) ? (_Float16)w[(long)oo * I + k] : (_Float16)0.0f;
}

// 3x3 OIHW [O][I][3][3] f32 -> [9][O][I] f16 (tap-major)
__global__ void prep_w3x3_kernel(const float* __restrict__ w, _Float16* __restrict__ o,
                                 int O, int I) {
  const long idx = (long)blockIdx.x * blockDim.x + threadIdx.x;
  const long tot = 9L * O * I;
  if (idx >= tot) return;
  const int i = (int)(idx % I);
  const int oo = (int)((idx / I) % O);
  const int tap = (int)(idx / ((long)I * O));
  o[idx] = (_Float16)w[((long)oo * I + i) * 9 + tap];
}

// ---------------------------------------------------------------------------
extern "C" void kernel_launch(void* const* d_in, const int* in_sizes, int n_in,
                              void* d_out, int out_size, void* d_ws, size_t ws_size,
                              hipStream_t stream) {
  (void)in_sizes; (void)n_in; (void)out_size; (void)ws_size;
  const int Himg = 256, Wimg = 256;
  const long P = 4L * Himg * Wimg;           // 262144 pixels
  const int PT = (int)(P / 16);              // 16384 pixel tiles
  const int PB = PT / 8;                     // 2048 blocks (8 waves/block)

  // -------- workspace layout (~334 MB) --------
  char* ws = (char*)d_ws;
  size_t off = 0;
  auto walloc = [&](size_t bytes) -> char* {
    char* pp = ws + off;
    off = (off + bytes + 255) & ~(size_t)255;
    return pp;
  };
  _Float16* X   = (_Float16*)walloc((size_t)P * 128 * 2);  // dense-stack concat buffer
  _Float16* A   = (_Float16*)walloc((size_t)P * 256 * 2);  // bn outputs / r1 out / f1 out
  _Float16* M   = (_Float16*)walloc((size_t)P * 128 * 2);  // main conv output
  _Float16* T   = (_Float16*)walloc((size_t)P * 112 * 2);  // d1a/d2a out, then filt [P][100]
  _Float16* wg  = (_Float16*)walloc((size_t)P * 4 * 2);    // space-to-depth of W
  float*    r2o = (float*)   walloc((size_t)P * 4 * 4);    // residual branch out
  _Float16* wT  = (_Float16*)walloc((size_t)9 * 128 * 128 * 2); // transposed f16 weights
  float*    st  = (float*)   walloc(256 * 4);              // sum[128] + sumsq[128]

  const float* bggr  = (const float*)d_in[0];
  const float* Wfull = (const float*)d_in[1];
  float* out = (float*)d_out;

  auto shb = [](int NB, int K) { return (size_t)NB * 16 * (K + 8) * 2; };

  // space-to-depth (once; reused by all 4 blocks)
  {
    long tot = P * 4;
    s2d_kernel<<<dim3((unsigned)((tot + 255) / 256)), dim3(256), 0, stream>>>(Wfull, wg, Himg, Wimg);
  }

  auto bn = [&](const _Float16* src, int ldIn, int C, _Float16* dst, int ldOut) {
    hipMemsetAsync(st, 0, 256 * 4, stream);
    bn_stats<<<dim3(512), dim3(128), 0, stream>>>(src, ldIn, C, P, st, st + 128);
    long tot = P * (long)C;
    bn_apply_relu<<<dim3((unsigned)((tot + 255) / 256)), dim3(256), 0, stream>>>(
        src, ldIn, C, P, st, st + 128, dst, ldOut);
  };
  auto prep1 = [&](const float* w, int O, int Opad, int I) {
    long tot = (long)Opad * I;
    prep_w1x1_kernel<<<dim3((unsigned)((tot + 255) / 256)), dim3(256), 0, stream>>>(w, wT, O, Opad, I);
  };
  auto prep3 = [&](const float* w, int O, int I) {
    long tot = 9L * O * I;
    prep_w3x3_kernel<<<dim3((unsigned)((tot + 255) / 256)), dim3(256), 0, stream>>>(w, wT, O, I);
  };

  for (int blk = 0; blk < 4; ++blk) {
#define PRM(k) ((const float*)d_in[2 + blk * 20 + (k)])
    const float *pre_w = PRM(0),  *pre_b = PRM(1);
    const float *d1a_w = PRM(2),  *d1a_b = PRM(3);
    const float *d1b_w = PRM(4),  *d1b_b = PRM(5);
    const float *d2a_w = PRM(6),  *d2a_b = PRM(7);
    const float *d2b_w = PRM(8),  *d2b_b = PRM(9);
    const float *main_w = PRM(10), *main_b = PRM(11);
    const float *r1_w = PRM(12),  *r1_b = PRM(13);
    const float *r2_w = PRM(14),  *r2_b = PRM(15);
    const float *f1_w = PRM(16),  *f1_b = PRM(17);
    const float *f2_w = PRM(18),  *f2_b = PRM(19);
#undef PRM

    // pre: 3x3 4->64 (direct) -> X[:,0:64]
    {
      long tot = P * 64;
      conv_pre_kernel<<<dim3((unsigned)((tot + 255) / 256)), dim3(256), 0, stream>>>(
          wg, pre_w, pre_b, X, Himg, Wimg);
    }
    // t = relu(bn(x)); d1a 1x1 64->64 (WMMA)
    bn(X, 128, 64, A, 64);
    prep1(d1a_w, 64, 64, 64);
    conv1x1_wmma<4, false, true><<<dim3(PB, 1), dim3(256), shb(4, 64), stream>>>(
        A, 64, 64, wT, d1a_b, 64, 64, T);
    // relu(bn); d1b 3x3 64->32 (WMMA) -> X[:,64:96]
    bn(T, 64, 64, A, 64);
    prep3(d1b_w, 32, 64);
    conv3x3_wmma<2, false, true><<<dim3(PB, 1), dim3(256), shb(2, 64), stream>>>(
        A, 64, 64, wT, d1b_b, 32, 32, 128, X + 64, Himg, Wimg);
    // relu(bn(x96)); d2a 1x1 96->96 (WMMA)
    bn(X, 128, 96, A, 96);
    prep1(d2a_w, 96, 96, 96);
    conv1x1_wmma<2, false, true><<<dim3(PB, 3), dim3(256), shb(2, 96), stream>>>(
        A, 96, 96, wT, d2a_b, 96, 96, T);
    // relu(bn); d2b 3x3 96->32 (WMMA) -> X[:,96:128]
    bn(T, 96, 96, A, 96);
    prep3(d2b_w, 32, 96);
    conv3x3_wmma<2, false, true><<<dim3(PB, 1), dim3(256), shb(2, 96), stream>>>(
        A, 96, 96, wT, d2b_b, 32, 32, 128, X + 96, Himg, Wimg);
    // relu(bn(x128)); main 3x3 128->128 + bias + relu (WMMA) -> M
    bn(X, 128, 128, A, 128);
    prep3(main_w, 128, 128);
    conv3x3_wmma<4, true, true><<<dim3(PB, 2), dim3(256), shb(4, 128), stream>>>(
        A, 128, 128, wT, main_b, 128, 128, 128, M, Himg, Wimg);
    // r branch: r1 1x1 128->128 + relu (WMMA) -> A ; r2 128->4 direct -> r2o
    prep1(r1_w, 128, 128, 128);
    conv1x1_wmma<4, true, true><<<dim3(PB, 2), dim3(256), shb(4, 128), stream>>>(
        M, 128, 128, wT, r1_b, 128, 128, A);
    {
      long tot = P * 4;
      conv_r2_kernel<<<dim3((unsigned)((tot + 255) / 256)), dim3(256), 0, stream>>>(
          A, r2_w, r2_b, r2o, P);
    }
    // f branch: f1 1x1 128->256 + relu (WMMA) -> A ; f2 256->100 (WMMA, padded to 112) -> T
    prep1(f1_w, 256, 256, 128);
    conv1x1_wmma<4, true, true><<<dim3(PB, 4), dim3(256), shb(4, 128), stream>>>(
        M, 128, 128, wT, f1_b, 256, 256, A);
    prep1(f2_w, 100, 112, 256);
    conv1x1_wmma<1, false, true><<<dim3(PB, 7), dim3(256), shb(1, 256), stream>>>(
        A, 256, 256, wT, f2_b, 100, 100, T);
    // softmax over 25 taps
    {
      long tot = P * 4;
      softmax25_kernel<<<dim3((unsigned)((tot + 255) / 256)), dim3(256), 0, stream>>>(T, P);
    }
    // dynamic filter + pixel shuffle + residual add -> out channel blk
    duf_out_kernel<<<dim3((unsigned)((P + 255) / 256)), dim3(256), 0, stream>>>(
        bggr, blk, T, r2o, out, Himg, Wimg);
  }
}
